// MultiHeadAttention_84129819394341
// MI455X (gfx1250) — matrix operand order repaired and checked by reading, not compiled, not run on hardware
//
#include <hip/hip_runtime.h>
#include <hip/hip_bf16.h>
#include <math.h>

// MHA for MI455X (gfx1250, wave32). f16 WMMA (V_WMMA_F32_16X16X32_F16) with
// f32 accumulation, LDS-tiled GEMMs, flash attention with async K-tile DMA
// (GLOBAL_LOAD_ASYNC_TO_LDS_B128, ASYNCcnt) and LDS-transposed V tiles.
//
// Fragment layouts per CDNA5 ISA 7.12.2:
//   A (16x32 f16): lane L holds row M=L%16; elems e -> K = e + (e&8) + 8*(L/16)
//   B (32x16 f16): lane L holds col N=L%16; elems e -> K = e + (e&8) + 8*(L/16)
//   C/D (16x16 f32): lane L, vgpr r -> M = r + 8*(L/16), N = L%16
//
// Shapes: B=4, S=2048, E=1024, H=16, Dk=64.  M = B*S = 8192.

typedef _Float16 v16h __attribute__((ext_vector_type(16)));
typedef _Float16 h8   __attribute__((ext_vector_type(8)));
typedef float    v8f  __attribute__((ext_vector_type(8)));
typedef float    f4   __attribute__((ext_vector_type(4)));

union Frag16 { v16h v; h8 h[2]; };

#define WMMA_F16(A_, B_, C_) \
  __builtin_amdgcn_wmma_f32_16x16x32_f16(false, (A_), false, (B_), (short)0, (C_), false, false)

#define MHA_E   1024
#define MHA_S   2048
#define MHA_B   4
#define MHA_H   16
#define MHA_DK  64
#define MHA_M   (MHA_B * MHA_S)             // 8192 rows
#define PROJ_ELEMS ((size_t)MHA_M * MHA_E)  // elements per f16 intermediate

// Async copy global->LDS, 16B per lane, tracked by ASYNCcnt (CDNA5 TDM-class path).
__device__ __forceinline__ void async_copy_b128(unsigned lds_off, const void* gaddr) {
  asm volatile("global_load_async_to_lds_b128 %0, %1, off"
               :: "v"(lds_off), "v"(gaddr) : "memory");
}
__device__ __forceinline__ void wait_asynccnt0() {
  asm volatile("s_wait_asynccnt 0" ::: "memory");
}

// ---------------------------------------------------------------------------
// GEMM:  Out[M,N] = A[M,K] * W[N,K]^T   (M=8192, N=K=1024)
// Block tile 128x128, K-step 32, staged through LDS as f16.
// 8 waves/block; wave tile 64x32 (4x2 C-fragments, 8 WMMAs per K-step).
// A_HALF:    A is f16 (ctx buffer) vs f32 (raw inputs)
// HEADS_OUT: write f16 in [B,H,S,Dk] (projections) vs f32 row-major (final)
// ---------------------------------------------------------------------------
template<bool A_HALF, bool HEADS_OUT>
__global__ __launch_bounds__(256)
void gemm_wmma_kernel(const void* __restrict__ Ap,
                      const float* __restrict__ W,
                      void* __restrict__ Outp)
{
  __shared__ __align__(16) _Float16 As[128 * 40];   // [row][k], stride 40
  __shared__ __align__(16) _Float16 Ws[128 * 40];   // [col][k], stride 40

  const int tid  = threadIdx.x;
  const int lane = tid & 31;
  const int wid  = tid >> 5;
  const int l16  = lane & 15;
  const int lh   = lane >> 4;
  const int wm   = wid >> 2;        // 0..1 -> 64-row slice
  const int wn   = wid & 3;         // 0..3 -> 32-col slice

  const int NB  = MHA_E / 128;      // 8 block-tiles along N
  const int m0b = (blockIdx.x / NB) * 128;
  const int n0b = (blockIdx.x % NB) * 128;

  const float*    Af = (const float*)Ap;
  const _Float16* Ah = (const _Float16*)Ap;

  // Staging: each thread owns one (row, 16-wide k half) of both slabs.
  const int srow = tid >> 1;          // 0..127
  const int sk   = (tid & 1) * 16;    // 0 or 16

  v8f c[4][2] = {};

  for (int k0 = 0; k0 < MHA_E; k0 += 32) {
    // ---- stage A slab (128x32) ----
    {
      _Float16* dst = &As[srow * 40 + sk];
      if (A_HALF) {
        const _Float16* g = Ah + (size_t)(m0b + srow) * MHA_E + k0 + sk;
        *(h8*)(dst)     = *(const h8*)(g);
        *(h8*)(dst + 8) = *(const h8*)(g + 8);
      } else {
        const float* g = Af + (size_t)(m0b + srow) * MHA_E + k0 + sk;
        f4 r0 = *(const f4*)(g);
        f4 r1 = *(const f4*)(g + 4);
        f4 r2 = *(const f4*)(g + 8);
        f4 r3 = *(const f4*)(g + 12);
        h8 x0, x1;
#pragma unroll
        for (int e = 0; e < 4; ++e) {
          x0[e] = (_Float16)r0[e]; x0[e + 4] = (_Float16)r1[e];
          x1[e] = (_Float16)r2[e]; x1[e + 4] = (_Float16)r3[e];
        }
        *(h8*)(dst)     = x0;
        *(h8*)(dst + 8) = x1;
        if (k0 + 32 < MHA_E)     // gfx1250 global_prefetch_b8 of next A slab
          __builtin_prefetch((const void*)(g + 32), 0, 3);
      }
    }
    // ---- stage W slab (128x32): B elem (k,n) = W[n*K + k] ----
    {
      const float* g = W + (size_t)(n0b + srow) * MHA_E + k0 + sk;
      f4 r0 = *(const f4*)(g);
      f4 r1 = *(const f4*)(g + 4);
      f4 r2 = *(const f4*)(g + 8);
      f4 r3 = *(const f4*)(g + 12);
      h8 x0, x1;
#pragma unroll
      for (int e = 0; e < 4; ++e) {
        x0[e] = (_Float16)r0[e]; x0[e + 4] = (_Float16)r1[e];
        x1[e] = (_Float16)r2[e]; x1[e + 4] = (_Float16)r3[e];
      }
      _Float16* dst = &Ws[srow * 40 + sk];
      *(h8*)(dst)     = x0;
      *(h8*)(dst + 8) = x1;
    }
    __syncthreads();

    // ---- fragments from LDS, 8 WMMAs ----
    Frag16 a[4], b[2];
#pragma unroll
    for (int i = 0; i < 4; ++i) {
      const int base = (wm * 64 + 16 * i + l16) * 40;
      a[i].h[0] = *(const h8*)&As[base + 8 * lh];
      a[i].h[1] = *(const h8*)&As[base + 16 + 8 * lh];
    }
#pragma unroll
    for (int j = 0; j < 2; ++j) {
      const int base = (wn * 32 + 16 * j + l16) * 40;
      b[j].h[0] = *(const h8*)&Ws[base + 8 * lh];
      b[j].h[1] = *(const h8*)&Ws[base + 16 + 8 * lh];
    }
#pragma unroll
    for (int i = 0; i < 4; ++i)
#pragma unroll
      for (int j = 0; j < 2; ++j)
        c[i][j] = WMMA_F16(a[i].v, b[j].v, c[i][j]);
    __syncthreads();
  }

  // ---- store (C layout: lane L vgpr r -> M = r + 8*lh, N = l16) ----
#pragma unroll
  for (int i = 0; i < 4; ++i)
#pragma unroll
    for (int j = 0; j < 2; ++j)
#pragma unroll
      for (int r = 0; r < 8; ++r) {
        const int Mg = m0b + wm * 64 + 16 * i + 8 * lh + r;
        const int Ng = n0b + wn * 32 + 16 * j + l16;
        if (HEADS_OUT) {
          // f16 output in [B, H, S, Dk]
          const int bb = Mg >> 11, s = Mg & (MHA_S - 1);
          const int h  = Ng >> 6,  d = Ng & (MHA_DK - 1);
          ((_Float16*)Outp)[(((size_t)(bb * MHA_H + h) * MHA_S + s) << 6) + d] =
              (_Float16)c[i][j][r];
        } else {
          ((float*)Outp)[(size_t)Mg * MHA_E + Ng] = c[i][j][r];
        }
      }
}

// ---------------------------------------------------------------------------
// Flash attention: one block (4 waves, 128 thr) per (b,h, 64-query block).
// Per 32-key step: K tile staged to LDS via async DMA (shared by all waves),
// V tile staged transposed to LDS, scores/PV via WMMA, online softmax with
// wave32 ds_bpermute reductions. P goes C-layout -> LDS -> A-layout.
// ---------------------------------------------------------------------------
__global__ __launch_bounds__(128)
void attn_kernel(const _Float16* __restrict__ qh,
                 const _Float16* __restrict__ kh,
                 const _Float16* __restrict__ vh,
                 _Float16* __restrict__ ch)
{
  __shared__ __align__(16) _Float16 kt[32 * 72];    // [key][d],  stride 72
  __shared__ __align__(16) _Float16 vt[64 * 40];    // [d][key],  stride 40 (transposed)
  __shared__ __align__(16) _Float16 plds[4 * 16 * 40];

  const int lane = threadIdx.x & 31;
  const int wid  = threadIdx.x >> 5;
  const int l16  = lane & 15;
  const int lh   = lane >> 4;

  const int bh = blockIdx.y;               // b*H + h
  const int q0 = blockIdx.x * 64 + wid * 16;

  const _Float16* qrow  = qh + ((size_t)bh * MHA_S + q0) * MHA_DK;
  const _Float16* kbase = kh + (size_t)bh * MHA_S * MHA_DK;
  const _Float16* vbase = vh + (size_t)bh * MHA_S * MHA_DK;
  _Float16* pw = plds + wid * 640;

  const unsigned kt_base = (unsigned)(size_t)&kt[0];

  // Preload Q A-fragments for d0 = 0, 32 (live in registers whole kernel)
  Frag16 aq[2];
#pragma unroll
  for (int t = 0; t < 2; ++t) {
    const int d0 = 32 * t;
    aq[t].h[0] = *(const h8*)(qrow + l16 * MHA_DK + d0 + 8 * lh);
    aq[t].h[1] = *(const h8*)(qrow + l16 * MHA_DK + d0 + 16 + 8 * lh);
  }

  v8f o[4] = {};
  float mrow[8], lrow[8];
#pragma unroll
  for (int r = 0; r < 8; ++r) { mrow[r] = -1e30f; lrow[r] = 0.0f; }

  for (int kb = 0; kb < MHA_S; kb += 32) {
    // ---- stage K tile (32x64 halfs) via async DMA: 256 16B chunks ----
#pragma unroll
    for (int cc0 = 0; cc0 < 2; ++cc0) {
      const int cc  = threadIdx.x + cc0 * 128;
      const int key = cc >> 3;
      const int c8  = (cc & 7) * 8;
      async_copy_b128(kt_base + (unsigned)(key * 72 + c8) * 2u,
                      (const void*)(kbase + (size_t)(kb + key) * MHA_DK + c8));
    }
    // ---- stage V tile transposed: vt[d][key] ----
    {
      const int key  = threadIdx.x >> 2;
      const int dblk = (threadIdx.x & 3) * 16;
      const _Float16* g = vbase + (size_t)(kb + key) * MHA_DK + dblk;
      h8 v0 = *(const h8*)(g);
      h8 v1 = *(const h8*)(g + 8);
#pragma unroll
      for (int e = 0; e < 8; ++e) {
        vt[(dblk + e) * 40 + key]     = v0[e];
        vt[(dblk + 8 + e) * 40 + key] = v1[e];
      }
    }
    wait_asynccnt0();          // our async chunks landed in LDS
    __syncthreads();           // all waves' stages visible

    // ---- scores: S = Q * K^T from LDS K tile ----
    v8f sc[2] = {};
#pragma unroll
    for (int t = 0; t < 2; ++t) {
      const int d0 = 32 * t;
#pragma unroll
      for (int j = 0; j < 2; ++j) {
        const int base = (16 * j + l16) * 72 + d0;
        Frag16 bk;
        bk.h[0] = *(const h8*)&kt[base + 8 * lh];
        bk.h[1] = *(const h8*)&kt[base + 16 + 8 * lh];
        sc[j] = WMMA_F16(aq[t].v, bk.v, sc[j]);
      }
    }

    // ---- online softmax (row stats replicated across each 16-lane half) ----
    float p[2][8], alpha[8];
#pragma unroll
    for (int r = 0; r < 8; ++r) {
      const float s0 = sc[0][r] * 0.125f;       // 1/sqrt(64)
      const float s1 = sc[1][r] * 0.125f;
      float v = fmaxf(s0, s1);
#pragma unroll
      for (int off = 1; off < 16; off <<= 1)
        v = fmaxf(v, __shfl_xor(v, off, 32));
      const float mn = fmaxf(mrow[r], v);
      alpha[r] = __expf(mrow[r] - mn);
      mrow[r]  = mn;
      p[0][r] = __expf(s0 - mn);
      p[1][r] = __expf(s1 - mn);
      float rs = p[0][r] + p[1][r];
#pragma unroll
      for (int off = 1; off < 16; off <<= 1)
        rs += __shfl_xor(rs, off, 32);
      lrow[r] = lrow[r] * alpha[r] + rs;
    }
#pragma unroll
    for (int d = 0; d < 4; ++d)
#pragma unroll
      for (int r = 0; r < 8; ++r)
        o[d][r] *= alpha[r];

    // ---- P: C-layout -> LDS -> A-layout ----
#pragma unroll
    for (int j = 0; j < 2; ++j)
#pragma unroll
      for (int r = 0; r < 8; ++r)
        pw[(8 * lh + r) * 40 + 16 * j + l16] = (_Float16)p[j][r];
    __syncthreads();

    Frag16 pa;
    pa.h[0] = *(const h8*)(pw + l16 * 40 + 8 * lh);
    pa.h[1] = *(const h8*)(pw + l16 * 40 + 16 + 8 * lh);

    // ---- O += P * V from transposed LDS V tile ----
#pragma unroll
    for (int j2 = 0; j2 < 4; ++j2) {
      const int base = (16 * j2 + l16) * 40;
      Frag16 bv;
      bv.h[0] = *(const h8*)&vt[base + 8 * lh];
      bv.h[1] = *(const h8*)&vt[base + 16 + 8 * lh];
      o[j2] = WMMA_F16(pa.v, bv.v, o[j2]);
    }
    __syncthreads();           // tiles free for next iteration's staging
  }

  // ---- normalize, store ctx as f16 in [B, S, H*Dk] = [8192, 1024] ----
  const int b = bh >> 4, h = bh & 15;
#pragma unroll
  for (int r = 0; r < 8; ++r) {
    const float inv = 1.0f / lrow[r];
    const int qg = q0 + 8 * lh + r;
    const size_t rowbase = ((size_t)(b * MHA_S + qg)) * MHA_E + h * MHA_DK;
#pragma unroll
    for (int j2 = 0; j2 < 4; ++j2)
      ch[rowbase + 16 * j2 + l16] = (_Float16)(o[j2][r] * inv);
  }
}

// ---------------------------------------------------------------------------
extern "C" void kernel_launch(void* const* d_in, const int* in_sizes, int n_in,
                              void* d_out, int out_size, void* d_ws, size_t ws_size,
                              hipStream_t stream)
{
  const float* Q  = (const float*)d_in[0];
  const float* K  = (const float*)d_in[1];
  const float* V  = (const float*)d_in[2];
  // d_in[3] = mask (unused by reference forward)
  const float* Wq = (const float*)d_in[4];
  const float* Wk = (const float*)d_in[5];
  const float* Wv = (const float*)d_in[6];
  const float* Wo = (const float*)d_in[7];

  // Workspace: 4 f16 buffers of 8192x1024 = 16 MiB each (64 MiB total).
  _Float16* qh = (_Float16*)d_ws;          // [B,H,S,Dk]
  _Float16* kh = qh + PROJ_ELEMS;
  _Float16* vh = kh + PROJ_ELEMS;
  _Float16* ch = vh + PROJ_ELEMS;          // ctx, [B,S,H*Dk]

  const int gemm_blocks = (MHA_M / 128) * (MHA_E / 128);   // 64*8 = 512

  gemm_wmma_kernel<false, true><<<gemm_blocks, 256, 0, stream>>>(Q, Wq, qh);
  gemm_wmma_kernel<false, true><<<gemm_blocks, 256, 0, stream>>>(K, Wk, kh);
  gemm_wmma_kernel<false, true><<<gemm_blocks, 256, 0, stream>>>(V, Wv, vh);

  attn_kernel<<<dim3(MHA_S / 64, MHA_B * MHA_H), 128, 0, stream>>>(qh, kh, vh, ch);

  gemm_wmma_kernel<true, false><<<gemm_blocks, 256, 0, stream>>>(ch, Wo, d_out);
}
